// Int8Linear_5540507811952
// MI455X (gfx1250) — 8-bit matrix operand order repaired and checked by reading, compiled, not run on hardware
//
#include <hip/hip_runtime.h>
#include <hip/hip_bf16.h>
#include <stdint.h>

typedef __attribute__((ext_vector_type(8))) int v8i;

#define TILE_M 128
#define TILE_N 128
#define TILE_K 64
#define LDS_STRIDE 80

static constexpr int Mdim = 8192;
static constexpr int Kdim = 4096;
static constexpr int Ndim = 4096;

__global__ void init_ws_kernel(unsigned int* absmax_bits) {
    if (threadIdx.x == 0) absmax_bits[0] = 0u;
}

__global__ __launch_bounds__(256) void absmax_kernel(const float4* __restrict__ w4,
                                                     unsigned int* __restrict__ absmax_bits,
                                                     int n4) {
    int tid = blockIdx.x * blockDim.x + threadIdx.x;
    int stride = gridDim.x * blockDim.x;
    float m = 0.0f;
    for (int i = tid; i < n4; i += stride) {
        float4 v = w4[i];
        m = fmaxf(m, fabsf(v.x));
        m = fmaxf(m, fabsf(v.y));
        m = fmaxf(m, fabsf(v.z));
        m = fmaxf(m, fabsf(v.w));
    }
    #pragma unroll
    for (int off = 16; off > 0; off >>= 1)
        m = fmaxf(m, __shfl_xor(m, off, 32));

    __shared__ float smax[8];
    int lane = threadIdx.x & 31;
    int wid  = threadIdx.x >> 5;
    if (lane == 0) smax[wid] = m;
    __syncthreads();
    if (threadIdx.x == 0) {
        float bm = smax[0];
        #pragma unroll
        for (int i = 1; i < 8; ++i) bm = fmaxf(bm, smax[i]);
        atomicMax(absmax_bits, __float_as_uint(bm));
    }
}

__device__ __forceinline__ int quant_pack4(float4 v, float rs) {
    int a = (int)rintf(fminf(fmaxf(v.x * rs, -127.0f), 127.0f));
    int b = (int)rintf(fminf(fmaxf(v.y * rs, -127.0f), 127.0f));
    int c = (int)rintf(fminf(fmaxf(v.z * rs, -127.0f), 127.0f));
    int d = (int)rintf(fminf(fmaxf(v.w * rs, -127.0f), 127.0f));
    return (a & 255) | ((b & 255) << 8) | ((c & 255) << 16) | ((d & 255) << 24);
}

__global__ __launch_bounds__(256) void quant_weight_kernel(const float4* __restrict__ w4,
                                                           int* __restrict__ wq,
                                                           const unsigned int* __restrict__ absmax_bits,
                                                           int n4) {
    int i = blockIdx.x * blockDim.x + threadIdx.x;
    if (i >= n4) return;
    float wscale = fmaxf(__uint_as_float(absmax_bits[0]) * (1.0f / 127.0f), 1e-8f);
    float rs = 1.0f / wscale;
    wq[i] = quant_pack4(w4[i], rs);
}

__global__ __launch_bounds__(256) void quant_x_kernel(const float4* __restrict__ x4,
                                                      int* __restrict__ xq,
                                                      const float* __restrict__ input_scale,
                                                      int n4) {
    int i = blockIdx.x * blockDim.x + threadIdx.x;
    if (i >= n4) return;
    float rs = 1.0f / input_scale[0];
    xq[i] = quant_pack4(x4[i], rs);
}

union AFrag { v8i v; int2 d[4]; };
union BFrag { v8i v; int4 q[2]; };

__global__ __launch_bounds__(256, 2) void int8_gemm_kernel(
    const signed char* __restrict__ Aq,
    const signed char* __restrict__ Bq,
    const float* __restrict__ bias,
    const float* __restrict__ input_scale,
    const unsigned int* __restrict__ absmax_bits,
    float* __restrict__ out) {

    __shared__ __align__(16) signed char lA[TILE_M * LDS_STRIDE];
    __shared__ __align__(16) signed char lB[TILE_N * LDS_STRIDE];

    const int tid  = threadIdx.x;
    const int lane = tid & 31;
    const int wid  = tid >> 5;

    const int bM = blockIdx.x * TILE_M;
    const int bN = blockIdx.y * TILE_N;

    const int waveM = (wid & 3) * 32;
    const int waveN = (wid >> 2) * 64;

    const int r      = lane & 15;
    const int half8  = (lane < 16) ? 0 : 8;
    const int half16 = (lane < 16) ? 0 : 16;

    v8i acc[2][4];
    #pragma unroll
    for (int tm = 0; tm < 2; ++tm)
        #pragma unroll
        for (int tn = 0; tn < 4; ++tn)
            acc[tm][tn] = (v8i)(0);

    for (int kb = 0; kb < Kdim; kb += TILE_K) {
        __syncthreads();
        #pragma unroll
        for (int c = 0; c < 2; ++c) {
            int id  = tid + c * 256;
            int row = id >> 2;
            int kc  = (id & 3) * 16;
            *(int4*)(&lA[row * LDS_STRIDE + kc]) =
                *(const int4*)(&Aq[(size_t)(bM + row) * Kdim + kb + kc]);
            *(int4*)(&lB[row * LDS_STRIDE + kc]) =
                *(const int4*)(&Bq[(size_t)(bN + row) * Kdim + kb + kc]);
        }
        if (kb + TILE_K < Kdim) {
            int id  = tid;
            int row = id >> 2;
            int kc  = (id & 3) * 16;
            __builtin_prefetch(&Aq[(size_t)(bM + row) * Kdim + kb + TILE_K + kc], 0, 3);
            __builtin_prefetch(&Bq[(size_t)(bN + row) * Kdim + kb + TILE_K + kc], 0, 3);
        }
        __syncthreads();

        AFrag a[2];
        #pragma unroll
        for (int tm = 0; tm < 2; ++tm) {
            const signed char* base = &lA[(waveM + tm * 16 + r) * LDS_STRIDE + half16];
            a[tm].d[0] = *(const int2*)(base + 0);
            a[tm].d[1] = *(const int2*)(base + 8);
            a[tm].d[2] = *(const int2*)(base + 32);
            a[tm].d[3] = *(const int2*)(base + 40);
        }
        BFrag b[4];
        #pragma unroll
        for (int tn = 0; tn < 4; ++tn) {
            const signed char* base = &lB[(waveN + tn * 16 + r) * LDS_STRIDE + half16];
            b[tn].q[0] = *(const int4*)(base + 0);
            b[tn].q[1] = *(const int4*)(base + 32);
        }

        #pragma unroll
        for (int tm = 0; tm < 2; ++tm)
            #pragma unroll
            for (int tn = 0; tn < 4; ++tn)
                acc[tm][tn] = __builtin_amdgcn_wmma_i32_16x16x64_iu8(
                     true, a[tm].v,
                     true, b[tn].v,
                    acc[tm][tn],
                     false,  false);
    }

    float wscale = fmaxf(__uint_as_float(absmax_bits[0]) * (1.0f / 127.0f), 1e-8f);
    float scale  = input_scale[0] * wscale;

    #pragma unroll
    for (int tm = 0; tm < 2; ++tm) {
        #pragma unroll
        for (int tn = 0; tn < 4; ++tn) {
            int col  = bN + waveN + tn * 16 + r;
            float bv = bias[col];
            #pragma unroll
            for (int j = 0; j < 8; ++j) {
                int row = bM + waveM + tm * 16 + j + half8;
                out[(size_t)row * Ndim + col] = (float)acc[tm][tn][j] * scale + bv;
            }
        }
    }
}

extern "C" void kernel_launch(void* const* d_in, const int* in_sizes, int n_in,
                              void* d_out, int out_size, void* d_ws, size_t ws_size,
                              hipStream_t stream) {
    const float* x           = (const float*)d_in[0];
    const float* weight      = (const float*)d_in[1];
    const float* bias        = (const float*)d_in[2];
    const float* input_scale = (const float*)d_in[3];

    float* out = (float*)d_out;

    unsigned int* absmax_bits = (unsigned int*)d_ws;
    signed char*  xq = (signed char*)d_ws + 256;
    signed char*  wq = xq + (size_t)Mdim * Kdim;

    init_ws_kernel<<<1, 1, 0, stream>>>(absmax_bits);

    const int w4 = (Ndim * Kdim) / 4;
    const int x4 = (Mdim * Kdim) / 4;

    absmax_kernel<<<1024, 256, 0, stream>>>((const float4*)weight, absmax_bits, w4);

    quant_weight_kernel<<<(w4 + 255) / 256, 256, 0, stream>>>(
        (const float4*)weight, (int*)wq, absmax_bits, w4);

    quant_x_kernel<<<(x4 + 255) / 256, 256, 0, stream>>>(
        (const float4*)x, (int*)xq, input_scale, x4);

    dim3 grid(Mdim / TILE_M, Ndim / TILE_N);
    int8_gemm_kernel<<<grid, 256, 0, stream>>>(xq, wq, bias, input_scale,
                                               absmax_bits, out);
}
